// LSTMWrapper_45157286150723
// MI455X (gfx1250) — matrix operand order repaired and checked by reading, not compiled, not run on hardware
//
#include <hip/hip_runtime.h>
#include <hip/hip_bf16.h>
#include <math.h>

#define BATCH 32
#define SEQ   512
#define INDIM 512
#define UNITS 1024
#define G4U   (4 * UNITS)          // 4096
#define MROWS (BATCH * SEQ)        // 16384
#define NWG   128                  // persistent workgroups in recurrence

typedef __attribute__((ext_vector_type(16))) __bf16 v16bf;
typedef __attribute__((ext_vector_type(8)))  __bf16 v8bf;
typedef __attribute__((ext_vector_type(8)))  float  v8f;
typedef __attribute__((ext_vector_type(4)))  float  vf4;

// workspace layout (bytes)
#define XP_BYTES ((size_t)MROWS * G4U * 2)            // bf16 x_proj, 128 MB
#define H_BYTES  ((size_t)2 * BATCH * UNITS * 2)      // bf16 h, double buffered
#define CNT_OFF  (XP_BYTES + H_BYTES)

// ---------------------------------------------------------------------------
// Init: zero h buffers and per-step barrier counters (runs every launch so
// graph replay is deterministic).
// ---------------------------------------------------------------------------
__global__ void lstm_init_kernel(unsigned* __restrict__ hwords,
                                 unsigned* __restrict__ cnt) {
  int i = blockIdx.x * blockDim.x + threadIdx.x;
  if (i < (int)(H_BYTES / 4)) hwords[i] = 0u;
  if (i < SEQ) cnt[i] = 0u;
}

// ---------------------------------------------------------------------------
// Phase 1: x_proj[m, n] = data[m, :] @ Wx[:, n] + b[n], stored bf16.
// Block = 128 threads (4 waves), 64x64 output tile, Wx tile staged in LDS.
// ---------------------------------------------------------------------------
__global__ __launch_bounds__(128)
void lstm_xproj_kernel(const float* __restrict__ data,
                       const float* __restrict__ Wx,
                       const float* __restrict__ bias,
                       __bf16* __restrict__ xp) {
  // 64 cols, col-major, padded stride 528 (1056 B = 33*32 -> 32B-aligned cols)
  __shared__ __bf16 WxS[64 * 528];

  const int tid = threadIdx.x;
  const int n0g = blockIdx.x * 64;
  const int m0g = blockIdx.y * 64;

  // Stage B tile: coalesced global reads (cols fast), fp32 -> bf16.
  for (int idx = tid; idx < 64 * 512; idx += 128) {
    int c = idx & 63, k = idx >> 6;
    WxS[c * 528 + k] = (__bf16)Wx[(size_t)k * G4U + n0g + c];
  }
  __syncthreads();

  const int lane = tid & 31;
  const int wid  = tid >> 5;
  const int mw = (wid & 1) * 32;      // wave row offset within 64x64 tile
  const int nw = (wid >> 1) * 32;     // wave col offset
  const int lr = lane & 15;
  const int hi = lane >> 4;

  v8f acc[2][2] = {};
  for (int kk = 0; kk < INDIM; kk += 32) {
    v16bf Af[2], Bf[2];
    for (int mt = 0; mt < 2; ++mt) {
      // A 16x32 bf16 layout: lane half 'hi' holds 16 contiguous K at [hi*16,+16)
      const float* ap = data + (size_t)(m0g + mw + mt * 16 + lr) * INDIM + kk + hi * 16;
      vf4 a0 = *(const vf4*)(ap);
      vf4 a1 = *(const vf4*)(ap + 4);
      vf4 a2 = *(const vf4*)(ap + 8);
      vf4 a3 = *(const vf4*)(ap + 12);
      v8bf lo, hiv;
      for (int i = 0; i < 4; ++i) {
        lo[i]      = (__bf16)a0[i];
        lo[4 + i]  = (__bf16)a1[i];
        hiv[i]     = (__bf16)a2[i];
        hiv[4 + i] = (__bf16)a3[i];
      }
      Af[mt] = __builtin_shufflevector(lo, hiv, 0, 1, 2, 3, 4, 5, 6, 7,
                                       8, 9, 10, 11, 12, 13, 14, 15);
    }
    for (int nt = 0; nt < 2; ++nt) {
      // B 32x16 bf16 layout: lane holds col (lane&15), 16 contiguous K at hi*16.
      // 32B-aligned -> single v16bf load (two ds_load_b128, no shuffles).
      Bf[nt] = *(const v16bf*)&WxS[(nw + nt * 16 + lr) * 528 + kk + hi * 16];
    }
    for (int mt = 0; mt < 2; ++mt)
      for (int nt = 0; nt < 2; ++nt)
        acc[mt][nt] = __builtin_amdgcn_wmma_f32_16x16x32_bf16(
            false, Af[mt], false, Bf[nt], (short)0, acc[mt][nt], false, false);
  }

  // Store D (+bias) as bf16: element v of v8f <-> row v + 8*hi, col = lane&15.
  for (int mt = 0; mt < 2; ++mt)
    for (int nt = 0; nt < 2; ++nt) {
      const int c = n0g + nw + nt * 16 + lr;
      const float bv = bias[c];
      for (int v = 0; v < 8; ++v) {
        const int r = m0g + mw + mt * 16 + v + 8 * hi;
        xp[(size_t)r * G4U + c] = (__bf16)(acc[mt][nt][v] + bv);
      }
    }
}

// ---------------------------------------------------------------------------
// Phase 2: persistent recurrence. 128 WGs x 256 threads (8 waves).
// Each WG owns 8 units (32 gate columns: i0..7,f0..7,g0..7,o0..7); its Wh
// slice lives in LDS (bf16, column-major) for all 512 steps. Cell state c is
// one register per thread. Steps separated by a device-scope barrier.
// ---------------------------------------------------------------------------
__global__ __launch_bounds__(256)
void lstm_rec_kernel(const float* __restrict__ Wh,
                     const __bf16* __restrict__ xp,
                     __bf16* __restrict__ hbuf,
                     unsigned* __restrict__ cnt,
                     float* __restrict__ out) {
  __shared__ __bf16 WhS[32 * 1024];   // 64 KB: 32 cols x 1024 K, col-major
  __shared__ float  gates[32 * 32];   // 4 KB: 32 batch x 32 cols

  const int tid = threadIdx.x;
  const int wg  = blockIdx.x;

  // Stage this WG's Wh slice once (fp32 -> bf16).
  for (int idx = tid; idx < 32 * 1024; idx += 256) {
    int c = idx & 31, k = idx >> 5;
    int gcol = (c >> 3) * UNITS + wg * 8 + (c & 7);
    WhS[c * 1024 + k] = (__bf16)Wh[(size_t)k * G4U + gcol];
  }
  __syncthreads();

  const int lane = tid & 31;
  const int wid  = tid >> 5;          // 0..7
  const int mt = wid & 1;             // batch-row tile (0..15 / 16..31)
  const int nt = (wid >> 1) & 1;      // col tile (0..15 / 16..31)
  const int kh = wid >> 2;            // K half (0: [0,512), 1: [512,1024))
  const int lr = lane & 15;
  const int hi = lane >> 4;

  const int eb = tid >> 3;            // elementwise: batch 0..31
  const int eu = tid & 7;             // elementwise: unit 0..7
  float cst = 0.0f;                   // cell state, lives in a register

  for (int t = 0; t < SEQ; ++t) {
    const __bf16* hc = hbuf + (size_t)(t & 1) * BATCH * UNITS;
    v8f acc = {};
    const int kbeg = kh * 512;
    for (int kk = kbeg; kk < kbeg + 512; kk += 32) {
      const __bf16* ap = hc + (size_t)(mt * 16 + lr) * UNITS + kk + hi * 16;
      v8bf a0 = *(const v8bf*)ap;
      v8bf a1 = *(const v8bf*)(ap + 8);
      v16bf A = __builtin_shufflevector(a0, a1, 0, 1, 2, 3, 4, 5, 6, 7,
                                        8, 9, 10, 11, 12, 13, 14, 15);
      // B: col-major LDS, 32B-aligned -> single v16bf load.
      v16bf B = *(const v16bf*)&WhS[(nt * 16 + lr) * 1024 + kk + hi * 16];
      acc = __builtin_amdgcn_wmma_f32_16x16x32_bf16(
          false, A, false, B, (short)0, acc, false, false);
    }

    // Reduce the two K halves through LDS.
    if (kh == 0)
      for (int v = 0; v < 8; ++v)
        gates[(mt * 16 + v + 8 * hi) * 32 + nt * 16 + lr] = acc[v];
    __syncthreads();
    if (kh == 1)
      for (int v = 0; v < 8; ++v)
        gates[(mt * 16 + v + 8 * hi) * 32 + nt * 16 + lr] += acc[v];
    __syncthreads();

    // Elementwise: one (batch, unit) per thread; i/f/g/o are local columns.
    const size_t xbase = ((size_t)eb * SEQ + t) * G4U + wg * 8 + eu;
    float gi = gates[eb * 32 + eu]       + (float)xp[xbase];
    float gf = gates[eb * 32 + 8 + eu]   + (float)xp[xbase + UNITS];
    float gg = gates[eb * 32 + 16 + eu]  + (float)xp[xbase + 2 * UNITS];
    float go = gates[eb * 32 + 24 + eu]  + (float)xp[xbase + 3 * UNITS];
    gi = 1.0f / (1.0f + __expf(-gi));
    gf = 1.0f / (1.0f + __expf(-gf));
    gg = tanhf(gg);
    go = 1.0f / (1.0f + __expf(-go));
    cst = gf * cst + gi * gg;
    const float hn = go * tanhf(cst);

    out[((size_t)eb * SEQ + t) * UNITS + wg * 8 + eu] = hn;
    __bf16* hnxt = hbuf + (size_t)((t + 1) & 1) * BATCH * UNITS;
    hnxt[eb * UNITS + wg * 8 + eu] = (__bf16)hn;

    // Device-scope barrier for step t: each wave release-signals (covers its
    // own h stores), one thread spins, then all acquire.
    if (lane == 0)
      __hip_atomic_fetch_add(&cnt[t], 1u, __ATOMIC_RELEASE, __HIP_MEMORY_SCOPE_AGENT);
    if (tid == 0) {
      while (__hip_atomic_load(&cnt[t], __ATOMIC_ACQUIRE, __HIP_MEMORY_SCOPE_AGENT)
             < (unsigned)(NWG * 8))
        __builtin_amdgcn_s_sleep(1);
    }
    __syncthreads();
    __builtin_amdgcn_fence(__ATOMIC_ACQUIRE, "agent");
  }
}

// ---------------------------------------------------------------------------
extern "C" void kernel_launch(void* const* d_in, const int* in_sizes, int n_in,
                              void* d_out, int out_size, void* d_ws, size_t ws_size,
                              hipStream_t stream) {
  (void)in_sizes; (void)n_in; (void)out_size; (void)ws_size;
  const float* data = (const float*)d_in[0];
  const float* Wx   = (const float*)d_in[1];
  const float* Wh   = (const float*)d_in[2];
  const float* bias = (const float*)d_in[3];
  float* out = (float*)d_out;

  char* ws = (char*)d_ws;
  __bf16*  xp   = (__bf16*)ws;
  __bf16*  hbuf = (__bf16*)(ws + XP_BYTES);
  unsigned* cnt = (unsigned*)(ws + CNT_OFF);

  lstm_init_kernel<<<(int)((H_BYTES / 4 + 255) / 256), 256, 0, stream>>>(
      (unsigned*)hbuf, cnt);
  lstm_xproj_kernel<<<dim3(G4U / 64, MROWS / 64), 128, 0, stream>>>(
      data, Wx, bias, xp);
  lstm_rec_kernel<<<NWG, 256, 0, stream>>>(Wh, xp, hbuf, cnt, out);
}
